// mfh_coatt_Med_59390807769379
// MI455X (gfx1250) — matrix operand order repaired and checked by reading, not compiled, not run on hardware
//
#include <hip/hip_runtime.h>
#include <math.h>

// ---- problem dims ----
#define BB   64
#define TT   20
#define DIN  200
#define HH   1024
#define G4H  (4*HH)
#define GQn  2
#define GIn  2
#define CCn  2048
#define SSn  196
#define OUTD 1000
#define FACn 5
#define JJn  5000
#define NOUTn 1700

typedef __attribute__((ext_vector_type(16))) _Float16 v16h;
typedef __attribute__((ext_vector_type(8)))  _Float16 v8h;
typedef __attribute__((ext_vector_type(8)))  float    v8f;

__device__ __forceinline__ float sigm(float x) { return 1.f / (1.f + expf(-x)); }
__device__ __forceinline__ float sgn_sqrt(float x) { return copysignf(sqrtf(fabsf(x)), x); }

// ============================================================================
// WMMA GEMM: D[M,N] = A[M,K](f32,row-major) x W[N,K](f32,row-major)^T + bias
// REQUIREMENT: M % 16 == 0 (true for every call site: 64 / 1280 / 12544).
//
// Block = 256 threads (8 waves) covering a 16(M) x 512(N) output tile.
// The 16-row A panel is shared by all waves: cooperatively staged into LDS
// (f32 -> f16, zero-padded past K) in KSEC-column sections using vectorized
// float4 loads + 16-byte ds stores; A fragments then come from ds_load_b128.
// Each wave computes a 16x64 tile: one A fragment feeds 4
// v_wmma_f32_16x16x32_f16 per 32-wide K chunk (f32 accumulate).
//
// EXEC discipline: no divergent control flow surrounds a WMMA. All waves run
// full compute (out-of-range waves use clamped W row pointers); only global
// stores are lane-predicated. The W K-tail is one separate chunk AFTER the
// hot loop so the hot loop has a single WMMA call site per accumulator.
// ============================================================================
#define KSEC    512              // staged K columns per section
#define ASTR    (KSEC + 8)       // padded f16 row stride (bank spread)

template<bool RELU>
__global__ __launch_bounds__(256) void gemm_wmma(
    const float* __restrict__ A, const float* __restrict__ W,
    const float* __restrict__ bias, float* __restrict__ D,
    int M, int N, int K)
{
  __shared__ _Float16 sA[16 * ASTR];       // 16.25 KB

  const int tid  = threadIdx.x;
  const int lane = tid & 31;
  const int wave = tid >> 5;
  const int l16  = lane & 15;
  const int half = lane >> 4;              // 0: lanes 0-15, 1: lanes 16-31
  const int m0   = blockIdx.y << 4;
  const int n0   = blockIdx.x * 512 + wave * 64;

  // W row pointers for the 4 column sub-tiles (B column = contiguous W row).
  // Clamped so out-of-range waves/lanes still make valid (harmless) loads.
  int nn[4], nc[4];
  const float* Wr[4];
  #pragma unroll
  for (int s = 0; s < 4; ++s) {
    nn[s] = n0 + 16 * s + l16;
    nc[s] = nn[s] < N ? nn[s] : N - 1;
    Wr[s] = W + (size_t)nc[s] * K;
  }

  // Lane k layout used here (wave32; contiguous, same order for A and B):
  //  A 16x32: lane holds row l%16, 16 contiguous K (offset 0/16 by lane half)
  //  B 32x16: lane holds col l%16, 16 contiguous K (offset 0/16 by lane half)
  const int abase = half << 4;             // 0 or 16
  const int wbase = half << 4;             // 0 or 16

  v8f acc[4] = {};

  for (int ks = 0; ks < K; ks += KSEC) {
    const int secLen = (K - ks) < KSEC ? (K - ks) : KSEC;

    // ---- cooperative stage of A panel section, 8 f16 per thread-iter:
    //      2x global float4 load -> cvt -> one 16-byte ds store ----
    __syncthreads();                       // previous section fully consumed
    for (int i8 = tid; i8 < 16 * (KSEC / 8); i8 += 256) {
      const int r  = i8 >> 6;              // / (KSEC/8)
      const int c0 = (i8 & 63) << 3;
      const float* src = A + (size_t)(m0 + r) * K + ks + c0;
      v8h hv;
      if (c0 + 8 <= secLen) {              // fast path: whole group in-bounds
        const float4 f0 = *(const float4*)(src);
        const float4 f1 = *(const float4*)(src + 4);
        hv[0] = (_Float16)f0.x; hv[1] = (_Float16)f0.y;
        hv[2] = (_Float16)f0.z; hv[3] = (_Float16)f0.w;
        hv[4] = (_Float16)f1.x; hv[5] = (_Float16)f1.y;
        hv[6] = (_Float16)f1.z; hv[7] = (_Float16)f1.w;
      } else {                             // K-tail straddle: zero-padded
        #pragma unroll
        for (int e = 0; e < 8; ++e)
          hv[e] = (_Float16)((c0 + e < secLen) ? src[e] : 0.f);
      }
      *(v8h*)&sA[r * ASTR + c0] = hv;
    }
    __syncthreads();

    // ---- hot loop: full 32-wide chunks only (single WMMA call site) ----
    const int fullEnd = secLen & ~31;
    for (int kc = 0; kc < fullEnd; kc += 32) {
      v16h av;
      const _Float16* ap = &sA[l16 * ASTR + kc + abase];
      #pragma unroll
      for (int e = 0; e < 8; ++e) { av[e] = ap[e]; av[e + 8] = ap[e + 8]; }
      #pragma unroll
      for (int s = 0; s < 4; ++s) {
        v16h bv;
        const float* wp = Wr[s] + ks + kc + wbase;
        #pragma unroll
        for (int e = 0; e < 16; ++e) bv[e] = (_Float16)wp[e];
        acc[s] = __builtin_amdgcn_wmma_f32_16x16x32_f16(
            false, av, false, bv, (short)0, acc[s], false, false);
      }
    }

    // ---- at most one K-tail chunk per section (W lane-predicated; A is
    //      zero-padded in LDS so it needs no predicates) ----
    if (fullEnd < secLen) {
      const int kc = fullEnd;
      v16h av;
      const _Float16* ap = &sA[l16 * ASTR + kc + abase];
      #pragma unroll
      for (int e = 0; e < 8; ++e) { av[e] = ap[e]; av[e + 8] = ap[e + 8]; }
      #pragma unroll
      for (int s = 0; s < 4; ++s) {
        v16h bv;
        #pragma unroll
        for (int e = 0; e < 16; ++e) {
          const int k = kc + wbase + e;                 // within section
          bv[e] = (_Float16)((k < secLen) ? Wr[s][ks + k] : 0.f);
        }
        acc[s] = __builtin_amdgcn_wmma_f32_16x16x32_f16(
            false, av, false, bv, (short)0, acc[s], false, false);
      }
    }
  }

  // ---- epilogue: D vgpr v, lane half h -> row m0 + v + 8h, col tile + l16 ----
  #pragma unroll
  for (int s = 0; s < 4; ++s) {
    const float bz = bias ? bias[nc[s]] : 0.f;
    #pragma unroll
    for (int v = 0; v < 8; ++v) {
      const int r = m0 + v + (half << 3);
      float x = acc[s][v] + bz;
      if (RELU) x = fmaxf(x, 0.f);
      if (nn[s] < N) D[(size_t)r * N + nn[s]] = x;
    }
  }
}

// ============================================================================
// Pointwise / reduction kernels
// ============================================================================
__global__ void lstm_cell(const float* __restrict__ Gx, const float* __restrict__ Gh,
                          const float* __restrict__ b_hh,
                          float* __restrict__ h, float* __restrict__ c,
                          float* __restrict__ hseq, int t)
{
  const int idx = blockIdx.x * blockDim.x + threadIdx.x;   // B*H
  if (idx >= BB * HH) return;
  const int b = idx / HH, j = idx % HH;
  const float* gx = Gx + ((size_t)(b * TT + t)) * G4H;     // includes b_ih
  const float* gh = Gh + (size_t)b * G4H;
  const float gi = gx[j]          + gh[j]          + b_hh[j];
  const float gf = gx[HH + j]     + gh[HH + j]     + b_hh[HH + j];
  const float gg = gx[2 * HH + j] + gh[2 * HH + j] + b_hh[2 * HH + j];
  const float go = gx[3 * HH + j] + gh[3 * HH + j] + b_hh[3 * HH + j];
  const float cn = sigm(gf) * c[idx] + sigm(gi) * tanhf(gg);
  const float hn = sigm(go) * tanhf(cn);
  c[idx] = cn;
  h[idx] = hn;
  hseq[((size_t)b * TT + t) * HH + j] = hn;
}

__global__ void qatt_logits(const float* __restrict__ qa, const float* __restrict__ Wq2,
                            const float* __restrict__ bq2, float* __restrict__ qlog)
{
  const int idx = blockIdx.x * blockDim.x + threadIdx.x;   // B*T*GQ
  if (idx >= BB * TT * GQn) return;
  const int g = idx % GQn, bt = idx / GQn;
  const float* ar = qa + (size_t)bt * 512;
  const float* wr = Wq2 + g * 512;
  float s = bq2[g];
  for (int k = 0; k < 512; ++k) s += ar[k] * wr[k];
  qlog[bt * GQn + g] = s;
}

__global__ void qatt_softmax(const float* __restrict__ qlog, float* __restrict__ qs)
{
  const int idx = blockIdx.x * blockDim.x + threadIdx.x;   // B*GQ
  if (idx >= BB * GQn) return;
  const int b = idx / GQn, g = idx % GQn;
  float mx = -1e30f;
  for (int t = 0; t < TT; ++t) mx = fmaxf(mx, qlog[(b * TT + t) * GQn + g]);
  float se = 0.f;
  for (int t = 0; t < TT; ++t) se += expf(qlog[(b * TT + t) * GQn + g] - mx);
  for (int t = 0; t < TT; ++t)
    qs[((size_t)b * GQn + g) * TT + t] = expf(qlog[(b * TT + t) * GQn + g] - mx) / se;
}

__global__ void qatt_feat(const float* __restrict__ qs, const float* __restrict__ hseq,
                          float* __restrict__ qfeat)
{
  const int idx = blockIdx.x * blockDim.x + threadIdx.x;   // B*GQ*H
  if (idx >= BB * GQn * HH) return;
  const int b = idx / (GQn * HH), g = (idx / HH) % GQn, j = idx % HH;
  const float* w = qs + ((size_t)b * GQn + g) * TT;
  float s = 0.f;
  for (int t = 0; t < TT; ++t) s += w[t] * hseq[((size_t)b * TT + t) * HH + j];
  qfeat[(size_t)b * (GQn * HH) + g * HH + j] = s;
}

__global__ void transpose_cs(const float* __restrict__ img, float* __restrict__ ifeatT)
{
  const int idx = blockIdx.x * blockDim.x + threadIdx.x;   // B*C*S
  if (idx >= BB * CCn * SSn) return;
  const int b = idx / (CCn * SSn), r = idx % (CCn * SSn);
  const int cc = r / SSn, s = r % SSn;
  ifeatT[((size_t)b * SSn + s) * CCn + cc] = img[idx];
}

__global__ void zpass(const float* __restrict__ iproj, const float* __restrict__ qproj,
                      float* __restrict__ zbuf, float* __restrict__ part)
{
  const int bs = blockIdx.x;                // B*S blocks
  const int b = bs / SSn;
  const float* ip = iproj + (size_t)bs * JJn;
  const float* qp = qproj + (size_t)b * JJn;
  float local = 0.f;
  for (int o = threadIdx.x; o < OUTD; o += blockDim.x) {
    float acc = 0.f;
    #pragma unroll
    for (int f = 0; f < FACn; ++f) { const int j = o * FACn + f; acc += qp[j] * ip[j]; }
    const float zv = sgn_sqrt(acc);
    zbuf[(size_t)bs * OUTD + o] = zv;
    local += zv * zv;
  }
  __shared__ float sh[256];
  sh[threadIdx.x] = local; __syncthreads();
  for (int st = 128; st > 0; st >>= 1) {
    if (threadIdx.x < st) sh[threadIdx.x] += sh[threadIdx.x + st];
    __syncthreads();
  }
  if (threadIdx.x == 0) part[bs] = sh[0];
}

__global__ void znorm(const float* __restrict__ part, float* __restrict__ invb)
{
  const int b = blockIdx.x;
  __shared__ float sh[256];
  float l = 0.f;
  for (int s = threadIdx.x; s < SSn; s += 256) l += part[b * SSn + s];
  sh[threadIdx.x] = l; __syncthreads();
  for (int st = 128; st > 0; st >>= 1) {
    if (threadIdx.x < st) sh[threadIdx.x] += sh[threadIdx.x + st];
    __syncthreads();
  }
  if (threadIdx.x == 0) invb[b] = 1.f / fmaxf(sqrtf(sh[0]), 1e-12f);
}

__global__ void zscale(float* __restrict__ zbuf, const float* __restrict__ invb)
{
  const int idx = blockIdx.x * blockDim.x + threadIdx.x;   // B*S*OUT
  if (idx >= BB * SSn * OUTD) return;
  zbuf[idx] *= invb[idx / (SSn * OUTD)];
}

__global__ void iatt_logits(const float* __restrict__ ia, const float* __restrict__ Wi2,
                            const float* __restrict__ bi2, float* __restrict__ ilog)
{
  const int idx = blockIdx.x * blockDim.x + threadIdx.x;   // B*S*GI
  if (idx >= BB * SSn * GIn) return;
  const int g = idx % GIn, bs = idx / GIn;
  const int b = bs / SSn, s = bs % SSn;
  const float* ar = ia + (size_t)bs * 512;
  const float* wr = Wi2 + g * 512;
  float v = bi2[g];
  for (int k = 0; k < 512; ++k) v += ar[k] * wr[k];
  ilog[((size_t)b * GIn + g) * SSn + s] = v;
}

__global__ void iatt_softmax(const float* __restrict__ ilog, float* __restrict__ isf)
{
  const int idx = blockIdx.x * blockDim.x + threadIdx.x;   // B*GI
  if (idx >= BB * GIn) return;
  const float* lr = ilog + (size_t)idx * SSn;
  float mx = -1e30f;
  for (int s = 0; s < SSn; ++s) mx = fmaxf(mx, lr[s]);
  float se = 0.f;
  for (int s = 0; s < SSn; ++s) se += expf(lr[s] - mx);
  for (int s = 0; s < SSn; ++s) isf[(size_t)idx * SSn + s] = expf(lr[s] - mx) / se;
}

__global__ void iatt_pool(const float* __restrict__ isf, const float* __restrict__ img,
                          float* __restrict__ iatt)
{
  const int idx = blockIdx.x * blockDim.x + threadIdx.x;   // B*GI*C
  if (idx >= BB * GIn * CCn) return;
  const int b = idx / (GIn * CCn), g = (idx / CCn) % GIn, cc = idx % CCn;
  const float* sf = isf + ((size_t)b * GIn + g) * SSn;
  const float* im = img + ((size_t)b * CCn + cc) * SSn;
  float s = 0.f;
  for (int t = 0; t < SSn; ++t) s += sf[t] * im[t];
  iatt[(size_t)b * (GIn * CCn) + g * CCn + cc] = s;
}

__global__ void pfuse(const float* __restrict__ o2q, const float* __restrict__ o2i,
                      const float* __restrict__ o3q, const float* __restrict__ o3i,
                      float* __restrict__ p2u, float* __restrict__ p3u)
{
  const int idx = blockIdx.x * blockDim.x + threadIdx.x;   // B*OUT
  if (idx >= BB * OUTD) return;
  const int b = idx / OUTD, o = idx % OUTD;
  const size_t base = (size_t)b * JJn;
  float s2 = 0.f, s3 = 0.f;
  #pragma unroll
  for (int f = 0; f < FACn; ++f) {
    const size_t j = base + o * FACn + f;
    const float v2 = o2q[j] * o2i[j];
    const float v3 = o3q[j] * o3i[j] * v2;
    s2 += v2; s3 += v3;
  }
  p2u[idx] = sgn_sqrt(s2);
  p3u[idx] = sgn_sqrt(s3);
}

__global__ void pnorm(const float* __restrict__ p2u, const float* __restrict__ p3u,
                      float* __restrict__ pf)
{
  const int b = blockIdx.x;
  __shared__ float s2[256], s3[256];
  float a = 0.f, c = 0.f;
  for (int o = threadIdx.x; o < OUTD; o += 256) {
    const float x = p2u[b * OUTD + o]; a += x * x;
    const float y = p3u[b * OUTD + o]; c += y * y;
  }
  s2[threadIdx.x] = a; s3[threadIdx.x] = c; __syncthreads();
  for (int st = 128; st > 0; st >>= 1) {
    if (threadIdx.x < st) { s2[threadIdx.x] += s2[threadIdx.x + st]; s3[threadIdx.x] += s3[threadIdx.x + st]; }
    __syncthreads();
  }
  const float i2 = 1.f / fmaxf(sqrtf(s2[0]), 1e-12f);
  const float i3 = 1.f / fmaxf(sqrtf(s3[0]), 1e-12f);
  for (int o = threadIdx.x; o < OUTD; o += 256) {
    pf[(size_t)b * 2 * OUTD + o]        = p2u[b * OUTD + o] * i2;
    pf[(size_t)b * 2 * OUTD + OUTD + o] = p3u[b * OUTD + o] * i3;
  }
}

__global__ void final_logsm(const float* __restrict__ logits, float* __restrict__ out)
{
  const int b = blockIdx.x;
  __shared__ float sh[256];
  const float* lr = logits + (size_t)b * NOUTn;
  float mx = -1e30f;
  for (int n = threadIdx.x; n < NOUTn; n += 256) mx = fmaxf(mx, sigm(lr[n]));
  sh[threadIdx.x] = mx; __syncthreads();
  for (int st = 128; st > 0; st >>= 1) {
    if (threadIdx.x < st) sh[threadIdx.x] = fmaxf(sh[threadIdx.x], sh[threadIdx.x + st]);
    __syncthreads();
  }
  mx = sh[0]; __syncthreads();
  float se = 0.f;
  for (int n = threadIdx.x; n < NOUTn; n += 256) se += expf(sigm(lr[n]) - mx);
  sh[threadIdx.x] = se; __syncthreads();
  for (int st = 128; st > 0; st >>= 1) {
    if (threadIdx.x < st) sh[threadIdx.x] += sh[threadIdx.x + st];
    __syncthreads();
  }
  const float lse = mx + logf(sh[0]);
  for (int n = threadIdx.x; n < NOUTn; n += 256) out[(size_t)b * NOUTn + n] = sigm(lr[n]) - lse;
}

// ============================================================================
// Host launcher
// ============================================================================
static inline void gemm(hipStream_t s, const float* A, const float* W, const float* bias,
                        float* D, int M, int N, int K, bool relu)
{
  dim3 grid((N + 511) / 512, (M + 15) / 16);   // M % 16 == 0 at all call sites
  if (relu) gemm_wmma<true ><<<grid, 256, 0, s>>>(A, W, bias, D, M, N, K);
  else      gemm_wmma<false><<<grid, 256, 0, s>>>(A, W, bias, D, M, N, K);
}

extern "C" void kernel_launch(void* const* d_in, const int* in_sizes, int n_in,
                              void* d_out, int out_size, void* d_ws, size_t ws_size,
                              hipStream_t stream)
{
  (void)in_sizes; (void)n_in; (void)out_size; (void)ws_size;
  const float* q_MED = (const float*)d_in[0];
  const float* img   = (const float*)d_in[1];
  const float* W_ih  = (const float*)d_in[2];
  const float* W_hh  = (const float*)d_in[3];
  const float* b_ih  = (const float*)d_in[4];
  const float* b_hh  = (const float*)d_in[5];
  const float* W1q = (const float*)d_in[6];  const float* b1q = (const float*)d_in[7];
  const float* W2q = (const float*)d_in[8];  const float* b2q = (const float*)d_in[9];
  const float* W3q = (const float*)d_in[10]; const float* b3q = (const float*)d_in[11];
  const float* Wci = (const float*)d_in[12]; const float* bci = (const float*)d_in[13];
  const float* W2i = (const float*)d_in[14]; const float* b2i = (const float*)d_in[15];
  const float* W3i = (const float*)d_in[16]; const float* b3i = (const float*)d_in[17];
  const float* Wq1 = (const float*)d_in[18]; const float* bq1 = (const float*)d_in[19];
  const float* Wq2 = (const float*)d_in[20]; const float* bq2 = (const float*)d_in[21];
  const float* Wi1 = (const float*)d_in[22]; const float* bi1 = (const float*)d_in[23];
  const float* Wi2 = (const float*)d_in[24]; const float* bi2 = (const float*)d_in[25];
  const float* Wp  = (const float*)d_in[26]; const float* bp  = (const float*)d_in[27];
  float* out = (float*)d_out;

  char* ws = (char*)d_ws; size_t off = 0;
  auto alloc = [&](size_t nelem) -> float* {
    float* p = (float*)(ws + off);
    off += ((nelem * sizeof(float)) + 255) & ~(size_t)255;
    return p;
  };
  float* Gx    = alloc((size_t)BB * TT * G4H);     // 21.0 MB
  float* Gh    = alloc((size_t)BB * G4H);
  float* hbuf  = alloc((size_t)BB * HH);
  float* cbuf  = alloc((size_t)BB * HH);
  float* hseq  = alloc((size_t)BB * TT * HH);      // [B,T,H]
  float* qa    = alloc((size_t)BB * TT * 512);
  float* qlog  = alloc((size_t)BB * TT * GQn);
  float* qs    = alloc((size_t)BB * GQn * TT);
  float* qfeat = alloc((size_t)BB * GQn * HH);     // [B,2048]
  float* qproj = alloc((size_t)BB * JJn);
  float* ifeatT= alloc((size_t)BB * SSn * CCn);    // 102.8 MB  [B*S, C]
  float* iproj = alloc((size_t)BB * SSn * JJn);    // 250.9 MB  [B*S, J]
  float* part  = alloc((size_t)BB * SSn);
  float* invb  = alloc((size_t)BB);
  float* ilog  = alloc((size_t)BB * GIn * SSn);
  float* isf   = alloc((size_t)BB * GIn * SSn);
  float* iatt  = alloc((size_t)BB * GIn * CCn);    // [B,4096]
  float* o2q   = alloc((size_t)BB * JJn);
  float* o2i   = alloc((size_t)BB * JJn);
  float* o3q   = alloc((size_t)BB * JJn);
  float* o3i   = alloc((size_t)BB * JJn);
  float* p2u   = alloc((size_t)BB * OUTD);
  float* p3u   = alloc((size_t)BB * OUTD);
  float* pf    = alloc((size_t)BB * 2 * OUTD);
  float* logits= alloc((size_t)BB * NOUTn);
  // aliased reuse (stream-ordered safe): ifeatT dead after iproj GEMM,
  // iproj dead after zpass.
  float* zbuf = ifeatT;   // [B*S, OUT]  50.2 MB <= 102.8 MB
  float* ia   = iproj;    // [B*S, 512]  25.7 MB <= 250.9 MB

  // --- LSTM: precompute x-projection for all (b,t), then 20 recurrent steps
  gemm(stream, q_MED, W_ih, b_ih, Gx, BB * TT, G4H, DIN, false);
  hipMemsetAsync(hbuf, 0, (size_t)BB * HH * sizeof(float), stream);
  hipMemsetAsync(cbuf, 0, (size_t)BB * HH * sizeof(float), stream);
  for (int t = 0; t < TT; ++t) {
    gemm(stream, hbuf, W_hh, nullptr, Gh, BB, G4H, HH, false);
    lstm_cell<<<(BB * HH + 255) / 256, 256, 0, stream>>>(Gx, Gh, b_hh, hbuf, cbuf, hseq, t);
  }

  // --- question attention
  gemm(stream, hseq, Wq1, bq1, qa, BB * TT, 512, HH, true);
  qatt_logits<<<(BB * TT * GQn + 255) / 256, 256, 0, stream>>>(qa, Wq2, bq2, qlog);
  qatt_softmax<<<1, BB * GQn, 0, stream>>>(qlog, qs);
  qatt_feat<<<(BB * GQn * HH + 255) / 256, 256, 0, stream>>>(qs, hseq, qfeat);

  // --- image attention with MFB
  transpose_cs<<<(BB * CCn * SSn + 255) / 256, 256, 0, stream>>>(img, ifeatT);
  gemm(stream, ifeatT, Wci, bci, iproj, BB * SSn, JJn, CCn, false);   // 257 GF GEMM
  gemm(stream, qfeat, W1q, b1q, qproj, BB, JJn, GQn * HH, false);
  zpass<<<BB * SSn, 256, 0, stream>>>(iproj, qproj, zbuf, part);
  znorm<<<BB, 256, 0, stream>>>(part, invb);
  zscale<<<(BB * SSn * OUTD + 255) / 256, 256, 0, stream>>>(zbuf, invb);
  gemm(stream, zbuf, Wi1, bi1, ia, BB * SSn, 512, OUTD, true);
  iatt_logits<<<(BB * SSn * GIn + 255) / 256, 256, 0, stream>>>(ia, Wi2, bi2, ilog);
  iatt_softmax<<<1, BB * GIn, 0, stream>>>(ilog, isf);
  iatt_pool<<<(BB * GIn * CCn + 255) / 256, 256, 0, stream>>>(isf, img, iatt);

  // --- MFH fusion (order-2 and order-3)
  gemm(stream, qfeat, W2q, b2q, o2q, BB, JJn, GQn * HH, false);
  gemm(stream, iatt,  W2i, b2i, o2i, BB, JJn, GIn * CCn, false);
  gemm(stream, qfeat, W3q, b3q, o3q, BB, JJn, GQn * HH, false);
  gemm(stream, iatt,  W3i, b3i, o3i, BB, JJn, GIn * CCn, false);
  pfuse<<<(BB * OUTD + 255) / 256, 256, 0, stream>>>(o2q, o2i, o3q, o3i, p2u, p3u);
  pnorm<<<BB, 256, 0, stream>>>(p2u, p3u, pf);

  // --- prediction head + log_softmax(sigmoid(.))
  gemm(stream, pf, Wp, bp, logits, BB, NOUTn, 2 * OUTD, false);
  final_logsm<<<BB, 256, 0, stream>>>(logits, out);
}
